// GPTLanguageModel_74071005987061
// MI455X (gfx1250) — compile-verified, matrix operand order checked by reading, not run on hardware
//
#include <hip/hip_runtime.h>
#include <hip/hip_bf16.h>

// ---------------- model constants ----------------
#define N_EMBD   1024
#define N_HEAD   16
#define HEAD_SZ  64
#define N_LAYER  8
#define SEQ      1024
#define VOCAB    32000
#define FFDIM    4096
#define BATCH    2
#define MROWS    (BATCH * SEQ)   // 2048

// ---------------- WMMA types ----------------
typedef __attribute__((ext_vector_type(16))) __bf16        v16bf;
typedef __attribute__((ext_vector_type(8)))  float         v8f;
typedef __attribute__((ext_vector_type(8)))  unsigned short ush8;
typedef unsigned short ush;

union ABFrag { v16bf v; ush8 h[2]; };

// native bf16 convert: let the compiler pick the hardware cvt instruction
__device__ __forceinline__ ush f2bf_u(float f) {
    __bf16 b = (__bf16)f;
    return __builtin_bit_cast(ush, b);
}

// =====================================================================
// Embedding: x[m][c] = tok_emb[idx[m]][c] + pos_emb[m % T][c]   (fp32 out)
// =====================================================================
__global__ void __launch_bounds__(256) embed_kernel(
    const int* __restrict__ idx, const float* __restrict__ tok,
    const float* __restrict__ pos, float* __restrict__ x)
{
    int i = blockIdx.x * 256 + threadIdx.x;          // over MROWS*C
    int m = i >> 10;
    int c = i & 1023;
    int t = m & (SEQ - 1);
    x[i] = tok[(size_t)idx[m] * N_EMBD + c] + pos[(size_t)t * N_EMBD + c];
}

// =====================================================================
// LayerNorm over C=1024, one block (256 threads, 8 waves) per row.
// fp32 in, bf16 out (feeds GEMM A operands).
// =====================================================================
__global__ void __launch_bounds__(256) layernorm_kernel(
    const float* __restrict__ x, const float* __restrict__ sc,
    const float* __restrict__ bi, ush* __restrict__ out)
{
    const int row = blockIdx.x;
    const int tid = threadIdx.x;
    const int lane = tid & 31, wid = tid >> 5;
    const float* xr = x + (size_t)row * N_EMBD;

    __shared__ float red[8];
    float v[4];
    float s = 0.f;
#pragma unroll
    for (int i = 0; i < 4; ++i) { v[i] = xr[tid + i * 256]; s += v[i]; }
#pragma unroll
    for (int o = 16; o >= 1; o >>= 1) s += __shfl_xor(s, o, 32);
    if (lane == 0) red[wid] = s;
    __syncthreads();
    float tot = 0.f;
#pragma unroll
    for (int w = 0; w < 8; ++w) tot += red[w];
    const float mu = tot * (1.0f / N_EMBD);

    float var = 0.f;
#pragma unroll
    for (int i = 0; i < 4; ++i) { float d = v[i] - mu; var += d * d; }
#pragma unroll
    for (int o = 16; o >= 1; o >>= 1) var += __shfl_xor(var, o, 32);
    __syncthreads();
    if (lane == 0) red[wid] = var;
    __syncthreads();
    tot = 0.f;
#pragma unroll
    for (int w = 0; w < 8; ++w) tot += red[w];
    const float rstd = rsqrtf(tot * (1.0f / N_EMBD) + 1e-5f);

    ush* orow = out + (size_t)row * N_EMBD;
#pragma unroll
    for (int i = 0; i < 4; ++i) {
        int c = tid + i * 256;
        orow[c] = f2bf_u((v[i] - mu) * rstd * sc[c] + bi[c]);
    }
}

// =====================================================================
// bf16-activation WMMA GEMM: C[M][N] = A[M][K](bf16) * W(k,n)(fp32->bf16)
//   BL=0: W row-major [K][N]
//   BL=1: QKV weight layout (H, K, 64): elem(k,n)=W[((n>>6)*K + k)*64 + (n&63)]
//   BL=2: W transposed [N][K]          (tied lm_head: tok_emb)
//   OBF : output bf16 (ush) instead of fp32
// block tile 128x128, 256 threads = 8 waves (2x4), wave tile 64x32.
// M,N multiples of 128; K multiple of 32 (true for all uses here).
// =====================================================================
template<int BL, bool BIAS, bool RELU, bool RES, bool OBF>
__global__ void __launch_bounds__(256) gemm_kernel(
    const ush* __restrict__ A, const float* __restrict__ W,
    const float* __restrict__ bias, const float* __restrict__ res,
    void* __restrict__ CoutV, int M, int N, int K)
{
    const int tid = threadIdx.x;
    const int bm = blockIdx.y * 128;
    const int bn = blockIdx.x * 128;
    const int wid = tid >> 5, lane = tid & 31;
    const int wr = wid >> 2, wc = wid & 3;          // 2 x 4 wave grid
    const int lc = lane & 15;
    const int kh = (lane >> 4) * 8;                 // per-lane K sub-chunk base

    __shared__ __align__(16) ush As[128][40];       // [m][k] bf16
    __shared__ __align__(16) ush Bs[128][40];       // [n][k] bf16

    v8f zero = {};
    v8f acc[4][2];
#pragma unroll
    for (int mt = 0; mt < 4; ++mt)
#pragma unroll
        for (int nt = 0; nt < 2; ++nt) acc[mt][nt] = zero;

    for (int k0 = 0; k0 < K; k0 += 32) {
        // ---- stage A tile (128 x 32): straight 2x16B bf16 copy ----
        {
            int row = tid >> 1;
            int col = (tid & 1) * 16;
            const ush* src = A + (size_t)(bm + row) * K + (k0 + col);
            if (k0 + 32 < K) __builtin_prefetch(src + 32, 0, 0);
            *(ush8*)&As[row][col]     = *(const ush8*)src;
            *(ush8*)&As[row][col + 8] = *(const ush8*)(src + 8);
        }
        // ---- stage B tile transposed -> Bs[n][k], fp32 -> bf16 ----
        if (BL == 2) {
            int n = tid >> 1;
            int c = (tid & 1) * 16;
            const float* src = W + (size_t)(bn + n) * K + (k0 + c);
#pragma unroll
            for (int j = 0; j < 16; ++j) Bs[n][c + j] = f2bf_u(src[j]);
        } else if (BL == 1) {
            int kr = tid >> 3;                       // 0..31
            int c  = (tid & 7) * 16;                 // 0..112, inside one head
            int n  = bn + c;
            const float* src = W + ((size_t)(n >> 6) * K + (k0 + kr)) * 64 + (n & 63);
#pragma unroll
            for (int j = 0; j < 16; ++j) Bs[c + j][kr] = f2bf_u(src[j]);
        } else {
            int kr = tid >> 3;
            int c  = (tid & 7) * 16;
            const float* src = W + (size_t)(k0 + kr) * N + (bn + c);
#pragma unroll
            for (int j = 0; j < 16; ++j) Bs[c + j][kr] = f2bf_u(src[j]);
        }
        __syncthreads();

        // ---- B fragments (reused across 4 m-tiles) ----
        ABFrag bf[2];
#pragma unroll
        for (int nt = 0; nt < 2; ++nt) {
            int n = wc * 32 + nt * 16 + lc;
            bf[nt].h[0] = *(const ush8*)&Bs[n][kh];
            bf[nt].h[1] = *(const ush8*)&Bs[n][16 + kh];
        }
        // ---- 8 WMMAs ----
#pragma unroll
        for (int mt = 0; mt < 4; ++mt) {
            ABFrag af;
            int m = wr * 64 + mt * 16 + lc;
            af.h[0] = *(const ush8*)&As[m][kh];
            af.h[1] = *(const ush8*)&As[m][16 + kh];
#pragma unroll
            for (int nt = 0; nt < 2; ++nt)
                acc[mt][nt] = __builtin_amdgcn_wmma_f32_16x16x32_bf16(
                    false, af.v, false, bf[nt].v, (short)0, acc[mt][nt], false, false);
        }
        __syncthreads();
    }

    // ---- epilogue (C layout: VGPR r = row r + 8*(lane>=16), col = lane&15) ----
    const int half8 = (lane >> 4) * 8;
#pragma unroll
    for (int mt = 0; mt < 4; ++mt)
#pragma unroll
        for (int nt = 0; nt < 2; ++nt)
#pragma unroll
            for (int r = 0; r < 8; ++r) {
                int row = bm + wr * 64 + mt * 16 + r + half8;
                int col = bn + wc * 32 + nt * 16 + lc;
                float vv = acc[mt][nt][r];
                if (BIAS) vv += bias[col];
                if (RES)  vv += res[(size_t)row * N + col];
                if (RELU) vv = fmaxf(vv, 0.0f);
                if (OBF)  ((ush*)CoutV)[(size_t)row * N + col] = f2bf_u(vv);
                else      ((float*)CoutV)[(size_t)row * N + col] = vv;
            }
}

// =====================================================================
// Flash attention (causal, online softmax) with WMMA; q/k/v/o are bf16.
// grid = (SEQ/128, BATCH*N_HEAD), 256 threads = 8 waves,
// each wave owns 16 query rows; key chunks of 32.
// layout: [(b*T + t) * C + h*64 + d]; 1/sqrt(D) applied post-S-WMMA.
// =====================================================================
__global__ void __launch_bounds__(256) attn_kernel(
    const ush* __restrict__ q, const ush* __restrict__ k,
    const ush* __restrict__ v, ush* __restrict__ o)
{
    const int bh = blockIdx.y;
    const int b  = bh >> 4;
    const int h  = bh & 15;
    const int wid  = threadIdx.x >> 5;
    const int lane = threadIdx.x & 31;
    const int q0   = blockIdx.x * 128 + wid * 16;
    const int half = lane >> 4;
    const int lc   = lane & 15;
    const int kh   = half * 8;

    __shared__ __align__(16) ush Pb[8][16][40];  // per-wave P (16x32) bf16
    __shared__ __align__(16) ush Vt[8][64][40];  // per-wave V^T (d x key) bf16

    const size_t base = ((size_t)b * SEQ) * N_EMBD + (size_t)h * HEAD_SZ;
    const ush* qb = q + base;
    const ush* kb = k + base;
    const ush* vb = v + base;
    ush*       ob = o + base;

    // Q fragments (K-dim = d = 64 -> 2 WMMA k-steps): direct 16B global loads
    ABFrag qf[2];
    {
        const ush* qrow = qb + (size_t)(q0 + lc) * N_EMBD;
#pragma unroll
        for (int s = 0; s < 2; ++s) {
            qf[s].h[0] = *(const ush8*)&qrow[s * 32 + kh];
            qf[s].h[1] = *(const ush8*)&qrow[s * 32 + 16 + kh];
        }
    }

    v8f zero = {};
    v8f oacc[4];
#pragma unroll
    for (int dt = 0; dt < 4; ++dt) oacc[dt] = zero;
    float mrow[8], lrow[8];
#pragma unroll
    for (int r = 0; r < 8; ++r) { mrow[r] = -3.0e38f; lrow[r] = 0.f; }

    for (int kk = 0; kk < q0 + 16; kk += 32) {
        // ---- S = q * k^T for 32 keys (two 16x16 C tiles) ----
        v8f sacc[2]; sacc[0] = zero; sacc[1] = zero;
#pragma unroll
        for (int nt = 0; nt < 2; ++nt) {
            const ush* krow = kb + (size_t)(kk + nt * 16 + lc) * N_EMBD;
#pragma unroll
            for (int s = 0; s < 2; ++s) {
                ABFrag kf;
                kf.h[0] = *(const ush8*)&krow[s * 32 + kh];
                kf.h[1] = *(const ush8*)&krow[s * 32 + 16 + kh];
                sacc[nt] = __builtin_amdgcn_wmma_f32_16x16x32_bf16(
                    false, qf[s].v, false, kf.v, (short)0, sacc[nt], false, false);
            }
        }

        // ---- scale + causal mask + online softmax ----
        float p0s[8], p1s[8];
#pragma unroll
        for (int r = 0; r < 8; ++r) {
            int qrow_i = q0 + r + half * 8;
            float s0 = sacc[0][r] * 0.125f;          // 1/sqrt(64)
            float s1 = sacc[1][r] * 0.125f;
            if (kk + lc > qrow_i)      s0 = -3.0e38f;
            if (kk + 16 + lc > qrow_i) s1 = -3.0e38f;
            float m = fmaxf(s0, s1);
#pragma unroll
            for (int off = 8; off >= 1; off >>= 1) m = fmaxf(m, __shfl_xor(m, off, 32));
            float newm  = fmaxf(mrow[r], m);
            float alpha = __expf(mrow[r] - newm);
            float p0 = __expf(s0 - newm);
            float p1 = __expf(s1 - newm);
            float ps = p0 + p1;
#pragma unroll
            for (int off = 8; off >= 1; off >>= 1) ps += __shfl_xor(ps, off, 32);
            lrow[r] = lrow[r] * alpha + ps;
            mrow[r] = newm;
#pragma unroll
            for (int dt = 0; dt < 4; ++dt) oacc[dt][r] *= alpha;
            p0s[r] = p0; p1s[r] = p1;
        }

        // ---- P -> LDS (re-layout C->A), V^T -> LDS (no conversion) ----
#pragma unroll
        for (int r = 0; r < 8; ++r) {
            Pb[wid][r + half * 8][lc]      = f2bf_u(p0s[r]);
            Pb[wid][r + half * 8][16 + lc] = f2bf_u(p1s[r]);
        }
        {
            const ush* vrow = vb + (size_t)(kk + lane) * N_EMBD;
#pragma unroll 8
            for (int d = 0; d < 64; ++d) Vt[wid][d][lane] = vrow[d];
        }
        __builtin_amdgcn_wave_barrier();

        // ---- O += P (16x32) * V (32x64) ----
        ABFrag pf;
        pf.h[0] = *(const ush8*)&Pb[wid][lc][kh];
        pf.h[1] = *(const ush8*)&Pb[wid][lc][16 + kh];
#pragma unroll
        for (int dt = 0; dt < 4; ++dt) {
            ABFrag vf;
            vf.h[0] = *(const ush8*)&Vt[wid][dt * 16 + lc][kh];
            vf.h[1] = *(const ush8*)&Vt[wid][dt * 16 + lc][16 + kh];
            oacc[dt] = __builtin_amdgcn_wmma_f32_16x16x32_bf16(
                false, pf.v, false, vf.v, (short)0, oacc[dt], false, false);
        }
        __builtin_amdgcn_wave_barrier();
    }

    // ---- normalize + store bf16 ----
#pragma unroll
    for (int r = 0; r < 8; ++r) {
        float inv = 1.0f / lrow[r];
        int row = q0 + r + half * 8;
        ush* orow = ob + (size_t)row * N_EMBD;
#pragma unroll
        for (int dt = 0; dt < 4; ++dt) orow[dt * 16 + lc] = f2bf_u(oacc[dt][r] * inv);
    }
}

// =====================================================================
// Host-side orchestration
// =====================================================================
extern "C" void kernel_launch(void* const* d_in, const int* in_sizes, int n_in,
                              void* d_out, int out_size, void* d_ws, size_t ws_size,
                              hipStream_t stream) {
    (void)in_sizes; (void)n_in; (void)out_size; (void)ws_size;

    const int*   idx    = (const int*)  d_in[0];
    const float* tok    = (const float*)d_in[1];
    const float* pos    = (const float*)d_in[2];
    const float* ln1_s  = (const float*)d_in[3];
    const float* ln1_b  = (const float*)d_in[4];
    const float* Wq     = (const float*)d_in[5];
    const float* Wk     = (const float*)d_in[6];
    const float* Wv     = (const float*)d_in[7];
    const float* Wo     = (const float*)d_in[8];
    const float* bo     = (const float*)d_in[9];
    const float* ln2_s  = (const float*)d_in[10];
    const float* ln2_b  = (const float*)d_in[11];
    const float* W1     = (const float*)d_in[12];
    const float* b1     = (const float*)d_in[13];
    const float* W2     = (const float*)d_in[14];
    const float* b2     = (const float*)d_in[15];
    const float* lnf_s  = (const float*)d_in[16];
    const float* lnf_b  = (const float*)d_in[17];
    float* out = (float*)d_out;

    const int C = N_EMBD, M = MROWS;
    const size_t MC = (size_t)M * C;

    // workspace layout (bytes): x fp32 | hb,qb,kb,vb,ao bf16 ; fb overlays qb..ao
    float* x  = (float*)d_ws;                    // MC fp32      (8 MB)
    ush*   hb = (ush*)(x + MC);                  // MC bf16      (4 MB)
    ush*   qb = hb + MC;                         // MC bf16
    ush*   kb = qb + MC;
    ush*   vb = kb + MC;
    ush*   ao = vb + MC;
    ush*   fb = qb;                              // M*FFDIM bf16 = 4*MC, covers qb..ao

    const dim3 blk(256);
    const dim3 gC (C      / 128, M / 128);   // N=1024
    const dim3 gFF(FFDIM  / 128, M / 128);   // N=4096
    const dim3 gV (VOCAB  / 128, M / 128);   // N=32000
    const dim3 gAtt(SEQ / 128, BATCH * N_HEAD);

    embed_kernel<<<(M * C) / 256, blk, 0, stream>>>(idx, tok, pos, x);

    for (int l = 0; l < N_LAYER; ++l) {
        const size_t oC  = (size_t)l * C;
        const size_t oCC = (size_t)l * C * C;        // also H*C*D
        const size_t oCF = (size_t)l * C * FFDIM;
        const size_t oF  = (size_t)l * FFDIM;

        layernorm_kernel<<<M, blk, 0, stream>>>(x, ln1_s + oC, ln1_b + oC, hb);

        gemm_kernel<1, false, false, false, true><<<gC, blk, 0, stream>>>(
            hb, Wq + oCC, nullptr, nullptr, qb, M, C, C);
        gemm_kernel<1, false, false, false, true><<<gC, blk, 0, stream>>>(
            hb, Wk + oCC, nullptr, nullptr, kb, M, C, C);
        gemm_kernel<1, false, false, false, true><<<gC, blk, 0, stream>>>(
            hb, Wv + oCC, nullptr, nullptr, vb, M, C, C);

        attn_kernel<<<gAtt, blk, 0, stream>>>(qb, kb, vb, ao);   // ao <- attn out (bf16)

        gemm_kernel<0, true, false, true, false><<<gC, blk, 0, stream>>>(
            ao, Wo + oCC, bo + oC, x, x, M, C, C);               // x += ao@Wo + bo

        layernorm_kernel<<<M, blk, 0, stream>>>(x, ln2_s + oC, ln2_b + oC, hb);

        gemm_kernel<0, true, true, false, true><<<gFF, blk, 0, stream>>>(
            hb, W1 + oCF, b1 + oF, nullptr, fb, M, FFDIM, C);    // fb = relu(hb@W1+b1)

        gemm_kernel<0, true, false, true, false><<<gC, blk, 0, stream>>>(
            fb, W2 + oCF, b2 + oC, x, x, M, C, FFDIM);           // x += fb@W2 + b2
    }

    layernorm_kernel<<<M, blk, 0, stream>>>(x, lnf_s, lnf_b, hb);

    gemm_kernel<2, false, false, false, false><<<gV, blk, 0, stream>>>(
        hb, tok, nullptr, nullptr, out, M, VOCAB, C);            // logits = hb @ tok^T
}
